// PETFullModel_197568496164
// MI455X (gfx1250) — compile-verified, matrix operand order checked by reading, not run on hardware
//
#include <hip/hip_runtime.h>
#include <math.h>

// ---------------------------------------------------------------------------
// PET full model for MI455X (gfx1250): bf16 WMMA GEMMs with async-LDS
// double-buffered B tiles, LDS WMMA attention.
// A=2048 atoms, N=64 neighbors, D=256, H=8 heads, DH=32, L=2 layers, V=119.
// ---------------------------------------------------------------------------

typedef __bf16 bf16_t;
typedef __attribute__((ext_vector_type(16))) __bf16 v16bf;
typedef __attribute__((ext_vector_type(8)))  __bf16 v8bf;
typedef __attribute__((ext_vector_type(8)))  float  v8f;

#define NATOM 2048
#define NNB   64
#define DM    256
#define NH    8
#define DH    32
#define NLAY  2
#define VSP   119
#define NTOK  (NATOM * NNB)   // 131072 tokens

__device__ __forceinline__ float gelu_f(float x) {
    float x3 = x * x * x;
    return 0.5f * x * (1.0f + tanhf(0.7978845608028654f * (x + 0.044715f * x3)));
}

// CDNA5 16-bit WMMA operand fragment: lane (l15 = row/col index, hi = lane>>4)
// holds 16 bf16 values: K = (j&7) + 8*hi + 16*(j>>3)  -> two contiguous 16B loads.
__device__ __forceinline__ v16bf load_frag(const bf16_t* rowptr, int k0, int hi) {
    const bf16_t* p = rowptr + k0 + 8 * hi;
    union { v16bf v; v8bf h[2]; } u;
    u.h[0] = *(const v8bf*)(p);
    u.h[1] = *(const v8bf*)(p + 16);
    return u.v;
}

// ---------------------------------------------------------------------------
// Generic bf16 WMMA GEMM:  out[M,N] = act(A[M,K] @ Bt[N,K]^T + bias (+ resid))
// Block = 8 waves stacked in M -> 128x64 block tile; wave computes 16x64 strip.
// The 64x32 B tile is shared by all 8 waves: staged once per K-step into LDS
// via GLOBAL_LOAD_ASYNC_TO_LDS_B128 (ASYNCcnt-tracked DMA), double-buffered so
// the next tile streams while WMMAs consume the current one.
// grid = (M/128, N/64).
// ---------------------------------------------------------------------------
template<int ACT, bool RESID, bool OUT_BF16>
__global__ __launch_bounds__(256)
void gemm_bf16_wmma(const bf16_t* __restrict__ Am, const bf16_t* __restrict__ Bt,
                    const float* __restrict__ bias, const float* __restrict__ resid,
                    float* __restrict__ outf, bf16_t* __restrict__ outh,
                    int M, int N, int K)
{
    __shared__ bf16_t btile[2][64 * 32];   // 4KB per buffer

    const int tid  = threadIdx.x;
    const int wave = tid >> 5;
    const int lane = tid & 31;
    const int l15 = lane & 15, hi = lane >> 4;
    const int m0 = (blockIdx.x * 8 + wave) * 16;
    const int n0 = blockIdx.y * 64;

    // cooperative B-tile copy: each of 256 threads owns one 16B chunk
    const int brow = tid >> 2, bpart = tid & 3;           // row 0..63, chunk 0..3
    const bf16_t* bsrc = Bt + (size_t)(n0 + brow) * K + bpart * 8;
    const unsigned lds_dst0 =
        (unsigned)(size_t)(&btile[0][0]) + (unsigned)(brow * 64 + bpart * 16);

    v8f acc[4] = {};
    const bf16_t* arow = Am + (size_t)(m0 + l15) * K;
    const int nsteps = K >> 5;

    {   // prefetch first B tile (async global -> LDS)
        uint64_t ga = (uint64_t)(size_t)bsrc;
        asm volatile("global_load_async_to_lds_b128 %0, %1, off"
                     :: "v"(lds_dst0), "v"(ga) : "memory");
    }
    for (int s = 0; s < nsteps; ++s) {
        if (s + 1 < nsteps) {
            uint64_t ga = (uint64_t)(size_t)(bsrc + (size_t)(s + 1) * 32);
            unsigned dst = lds_dst0 + (unsigned)(((s + 1) & 1) * 4096);
            asm volatile("global_load_async_to_lds_b128 %0, %1, off"
                         :: "v"(dst), "v"(ga) : "memory");
            // two tiles in flight; async loads complete in order -> cnt<=1
            // guarantees tile s has landed in LDS.
            asm volatile("s_wait_asynccnt 0x1" ::: "memory");
            __builtin_prefetch(arow + (size_t)(s + 1) * 32, 0, 1);
        } else {
            asm volatile("s_wait_asynccnt 0x0" ::: "memory");
        }
        __syncthreads();   // all waves' async tiles visible before ds reads

        v16bf a = load_frag(arow, s * 32, hi);
        const bf16_t* bb = &btile[s & 1][0];
#pragma unroll
        for (int t = 0; t < 4; ++t) {
            v16bf b = load_frag(bb + (t * 16 + l15) * 32, 0, hi);
            acc[t] = __builtin_amdgcn_wmma_f32_16x16x32_bf16(
                false, a, false, b, (short)0, acc[t], false, false);
        }
        __syncthreads();   // ds reads done before buffer is overwritten
    }

    // C layout: lanes 0-15: vgpr i -> row m0+i ; lanes 16-31: row m0+8+i
#pragma unroll
    for (int t = 0; t < 4; ++t) {
#pragma unroll
        for (int i = 0; i < 8; ++i) {
            int r = m0 + i + 8 * hi;
            int c = n0 + t * 16 + l15;
            float v = acc[t][i] + bias[c];
            if (RESID) v += resid[(size_t)r * N + c];
            if (ACT == 1) v = gelu_f(v);
            if (OUT_BF16) outh[(size_t)r * N + c] = (bf16_t)v;
            else          outf[(size_t)r * N + c] = v;
        }
    }
}

// ---------------------------------------------------------------------------
// Attention: block = (atom, half) -> 4 heads, 2 waves per head.
// Q,K,Vt staged bf16 in LDS (48KB/block). scores = Q@K^T (1 WMMA per 16x16
// tile, K-dim = DH = 32), masked softmax with shfl reductions, probs -> LDS
// (aliasing Q/K region), attn = probs@V via WMMA over 2 K-steps.
// ---------------------------------------------------------------------------
__global__ __launch_bounds__(256)
void attention_kernel(const float* __restrict__ qkv,          // [NTOK, 768]
                      const unsigned char* __restrict__ mask, // [A, N]
                      const float* __restrict__ cutoff,       // [A, N]
                      bf16_t* __restrict__ attn_out)          // [NTOK, 256]
{
    __shared__ bf16_t sm[4][3 * 64 * 32];   // per head-slot: Qs|Ks|Vt (12KB)
    const int a    = blockIdx.x >> 1;
    const int half = blockIdx.x & 1;
    const int wave = threadIdx.x >> 5;
    const int lane = threadIdx.x & 31;
    const int l15 = lane & 15, hi = lane >> 4;
    const int slot = wave >> 1;            // 0..3
    const int head = half * 4 + slot;
    const int sub  = wave & 1;             // which row-half of this head
    bf16_t* Qs = sm[slot];
    bf16_t* Ks = Qs + 64 * 32;
    bf16_t* Vt = Qs + 2 * 64 * 32;
    bf16_t* Ps = Qs;                       // 64x64 probs aliases Qs+Ks
    const float invs = 0.17677669529663687f;  // 1/sqrt(DH)

    // cooperative load of this head's Q,K,V (2 waves = 64 lanes)
    const float* base = qkv + (size_t)a * 64 * 768 + head * 32;
    for (int idx = (threadIdx.x & 63); idx < 2048; idx += 64) {
        int row = idx >> 5, col = idx & 31;
        const float* rp = base + (size_t)row * 768;
        Qs[row * 32 + col] = (bf16_t)rp[col];
        Ks[row * 32 + col] = (bf16_t)rp[256 + col];
        Vt[col * 64 + row] = (bf16_t)rp[512 + col];
    }
    __syncthreads();

    // scores: this wave owns row tiles it = sub*2 .. sub*2+1
    v8f sc[2][4] = {};
#pragma unroll
    for (int itl = 0; itl < 2; ++itl) {
        int it = sub * 2 + itl;
        v16bf af = load_frag(Qs + (it * 16 + l15) * 32, 0, hi);
#pragma unroll
        for (int jt = 0; jt < 4; ++jt) {
            v16bf bfr = load_frag(Ks + (jt * 16 + l15) * 32, 0, hi);
            sc[itl][jt] = __builtin_amdgcn_wmma_f32_16x16x32_bf16(
                false, af, false, bfr, (short)0, sc[itl][jt], false, false);
        }
    }
    __syncthreads();   // both waves of head done reading Qs/Ks before Ps overwrite

    bool mk[4]; float cf[4];
#pragma unroll
    for (int jt = 0; jt < 4; ++jt) {
        int col = jt * 16 + l15;
        mk[jt] = mask[(size_t)a * NNB + col] != 0;
        cf[jt] = cutoff[(size_t)a * NNB + col];
    }
    // softmax: row = it*16 + i + 8*hi lives across the 16 lanes of this half
#pragma unroll
    for (int itl = 0; itl < 2; ++itl) {
#pragma unroll
        for (int i = 0; i < 8; ++i) {
            float v[4];
#pragma unroll
            for (int jt = 0; jt < 4; ++jt) {
                float s = sc[itl][jt][i] * invs;
                v[jt] = mk[jt] ? -1e9f : s;
            }
            float m = fmaxf(fmaxf(v[0], v[1]), fmaxf(v[2], v[3]));
#pragma unroll
            for (int s = 1; s < 16; s <<= 1) m = fmaxf(m, __shfl_xor(m, s, 32));
            float p[4], sum = 0.f;
#pragma unroll
            for (int jt = 0; jt < 4; ++jt) { p[jt] = __expf(v[jt] - m); sum += p[jt]; }
#pragma unroll
            for (int s = 1; s < 16; s <<= 1) sum += __shfl_xor(sum, s, 32);
            float rinv = 1.0f / sum;
            int row = (sub * 2 + itl) * 16 + i + 8 * hi;
#pragma unroll
            for (int jt = 0; jt < 4; ++jt)
                Ps[row * 64 + jt * 16 + l15] = (bf16_t)(p[jt] * rinv * cf[jt]);
        }
    }
    __syncthreads();

    // attn = probs @ V : K-dim 64 -> 2 WMMA steps; N-dim 32 -> 2 col tiles
    v8f oacc[2][2] = {};
#pragma unroll
    for (int kk = 0; kk < 2; ++kk) {
#pragma unroll
        for (int itl = 0; itl < 2; ++itl) {
            int it = sub * 2 + itl;
            v16bf af = load_frag(Ps + (it * 16 + l15) * 64, kk * 32, hi);
#pragma unroll
            for (int nt = 0; nt < 2; ++nt) {
                v16bf bfr = load_frag(Vt + (nt * 16 + l15) * 64, kk * 32, hi);
                oacc[itl][nt] = __builtin_amdgcn_wmma_f32_16x16x32_bf16(
                    false, af, false, bfr, (short)0, oacc[itl][nt], false, false);
            }
        }
    }
#pragma unroll
    for (int itl = 0; itl < 2; ++itl) {
        int it = sub * 2 + itl;
#pragma unroll
        for (int nt = 0; nt < 2; ++nt) {
#pragma unroll
            for (int i = 0; i < 8; ++i) {
                int tok = a * 64 + it * 16 + i + 8 * hi;
                attn_out[(size_t)tok * DM + head * DH + nt * 16 + l15] =
                    (bf16_t)oacc[itl][nt][i];
            }
        }
    }
}

// ---------------------------------------------------------------------------
// Elementwise / reduction kernels
// ---------------------------------------------------------------------------
__global__ __launch_bounds__(256)
void zero_out_kernel(float* out, int n) {
    int i = blockIdx.x * 256 + threadIdx.x;
    if (i < n) out[i] = 0.f;
}

__global__ __launch_bounds__(256)
void msgs_init_kernel(const int* __restrict__ nb_species,
                      const float* __restrict__ neigh_embed,
                      float* __restrict__ msgs)
{
    size_t idx = (size_t)blockIdx.x * 256 + threadIdx.x;   // NTOK*DM
    int t = (int)(idx >> 8), d = (int)(idx & 255);
    msgs[idx] = neigh_embed[(size_t)nb_species[t] * DM + d];
}

__global__ __launch_bounds__(256)
void nodes_gather_kernel(const int* __restrict__ species,
                         const float* __restrict__ node_emb_l,
                         float* __restrict__ nodes)
{
    size_t idx = (size_t)blockIdx.x * 256 + threadIdx.x;   // NATOM*DM
    int a = (int)(idx >> 8), d = (int)(idx & 255);
    nodes[idx] = node_emb_l[(size_t)species[a] * DM + d];
}

// x = msgs + gelu(efeat@We + be) + nodes ; h = LN1(x) as bf16. One block/token.
__global__ __launch_bounds__(256)
void build_x_ln1_kernel(const float* __restrict__ msgs,
                        const float* __restrict__ evec,
                        const float* __restrict__ edist,
                        const int* __restrict__ species,
                        const float* __restrict__ node_emb_l,
                        const float* __restrict__ We, const float* __restrict__ be,
                        const float* __restrict__ g, const float* __restrict__ b,
                        float* __restrict__ x, bf16_t* __restrict__ h)
{
    __shared__ float red[256];
    int t = blockIdx.x, d = threadIdx.x;
    int a = t >> 6;
    float e = evec[(size_t)t * 3 + 0] * We[d]
            + evec[(size_t)t * 3 + 1] * We[DM + d]
            + evec[(size_t)t * 3 + 2] * We[2 * DM + d]
            + edist[t]                * We[3 * DM + d] + be[d];
    float xv = msgs[(size_t)t * DM + d] + gelu_f(e)
             + node_emb_l[(size_t)species[a] * DM + d];
    x[(size_t)t * DM + d] = xv;

    red[d] = xv; __syncthreads();
    for (int s = 128; s > 0; s >>= 1) { if (d < s) red[d] += red[d + s]; __syncthreads(); }
    float mean = red[0] * (1.f / DM); __syncthreads();
    float c = xv - mean;
    red[d] = c * c; __syncthreads();
    for (int s = 128; s > 0; s >>= 1) { if (d < s) red[d] += red[d + s]; __syncthreads(); }
    float r = rsqrtf(red[0] * (1.f / DM) + 1e-5f);
    h[(size_t)t * DM + d] = (bf16_t)(c * r * g[d] + b[d]);
}

__global__ __launch_bounds__(256)
void ln_kernel(const float* __restrict__ x, const float* __restrict__ g,
               const float* __restrict__ b, bf16_t* __restrict__ h)
{
    __shared__ float red[256];
    int t = blockIdx.x, d = threadIdx.x;
    float xv = x[(size_t)t * DM + d];
    red[d] = xv; __syncthreads();
    for (int s = 128; s > 0; s >>= 1) { if (d < s) red[d] += red[d + s]; __syncthreads(); }
    float mean = red[0] * (1.f / DM); __syncthreads();
    float c = xv - mean;
    red[d] = c * c; __syncthreads();
    for (int s = 128; s > 0; s >>= 1) { if (d < s) red[d] += red[d + s]; __syncthreads(); }
    float r = rsqrtf(red[0] * (1.f / DM) + 1e-5f);
    h[(size_t)t * DM + d] = (bf16_t)(c * r * g[d] + b[d]);
}

// output_edge = mask?0:x (zero x in place, bf16 copy), pooled = sum_n oe*cutoff
__global__ __launch_bounds__(256)
void mask_pool_kernel(float* __restrict__ x, const unsigned char* __restrict__ mask,
                      const float* __restrict__ cutoff,
                      bf16_t* __restrict__ oe, bf16_t* __restrict__ pooled)
{
    int a = blockIdx.x, d = threadIdx.x;
    float s = 0.f;
    for (int n = 0; n < NNB; ++n) {
        size_t idx = ((size_t)a * NNB + n) * DM + d;
        float xv = x[idx];
        bool mk = mask[(size_t)a * NNB + n] != 0;
        if (mk) { xv = 0.f; x[idx] = 0.f; }
        oe[idx] = (bf16_t)xv;
        s += xv * (mk ? 0.f : cutoff[(size_t)a * NNB + n]);
    }
    pooled[(size_t)a * DM + d] = (bf16_t)s;
}

__global__ __launch_bounds__(256)
void f32_to_bf16_kernel(const float* __restrict__ in, bf16_t* __restrict__ out, int n) {
    int i = blockIdx.x * 256 + threadIdx.x;
    if (i < n) out[i] = (bf16_t)in[i];
}

// f32 [K,N] -> bf16 [N,K] (so GEMM B-fragments load contiguous K)
__global__ __launch_bounds__(256)
void transpose_to_bf16_kernel(const float* __restrict__ in, bf16_t* __restrict__ out,
                              int K, int N)
{
    int idx = blockIdx.x * 256 + threadIdx.x;
    if (idx >= K * N) return;
    int k = idx / N, n = idx - k * N;
    out[(size_t)n * K + k] = (bf16_t)in[idx];
}

// node_e[a] = dot(nodeh[a,:], Wnf) + bnf  -> atomicAdd into energies
__global__ __launch_bounds__(256)
void node_energy_kernel(const bf16_t* __restrict__ nodeh, const float* __restrict__ Wnf,
                        const float* __restrict__ bnf, float* __restrict__ out)
{
    int wave = threadIdx.x >> 5, lane = threadIdx.x & 31;
    int a = blockIdx.x * 8 + wave;
    float s = 0.f;
#pragma unroll
    for (int j = 0; j < 8; ++j) {
        int d = lane * 8 + j;
        s += (float)nodeh[(size_t)a * DM + d] * Wnf[d];
    }
    for (int m = 16; m > 0; m >>= 1) s += __shfl_xor(s, m, 32);
    if (lane == 0) atomicAdd(&out[a], s + bnf[0]);
}

// edge_e[t] = (dot(eh[t,:],Wef)+bef) masked * cutoff -> atomicAdd into energies
__global__ __launch_bounds__(256)
void edge_energy_kernel(const bf16_t* __restrict__ eh, const float* __restrict__ Wef,
                        const float* __restrict__ bef,
                        const unsigned char* __restrict__ mask,
                        const float* __restrict__ cutoff, float* __restrict__ out)
{
    int wave = threadIdx.x >> 5, lane = threadIdx.x & 31;
    int t = blockIdx.x * 8 + wave;
    int a = t >> 6;
    float s = 0.f;
#pragma unroll
    for (int j = 0; j < 8; ++j) {
        int d = lane * 8 + j;
        s += (float)eh[(size_t)t * DM + d] * Wef[d];
    }
    for (int m = 16; m > 0; m >>= 1) s += __shfl_xor(s, m, 32);
    if (lane == 0 && mask[t] == 0) atomicAdd(&out[a], (s + bef[0]) * cutoff[t]);
}

// msgs = 0.5*(msgs + masked gather of output_edge (x) via reverse index)
__global__ __launch_bounds__(256)
void rev_mix_kernel(float* __restrict__ msgs, const float* __restrict__ x,
                    const int* __restrict__ rev, const unsigned char* __restrict__ mask)
{
    size_t idx = (size_t)blockIdx.x * 256 + threadIdx.x;  // NTOK*DM
    int t = (int)(idx >> 8), d = (int)(idx & 255);
    float v = mask[t] ? 0.f : x[(size_t)rev[t] * DM + d];
    msgs[idx] = 0.5f * (msgs[idx] + v);
}

// ---------------------------------------------------------------------------
// Host orchestration
// ---------------------------------------------------------------------------
extern "C" void kernel_launch(void* const* d_in, const int* in_sizes, int n_in,
                              void* d_out, int out_size, void* d_ws, size_t ws_size,
                              hipStream_t stream)
{
    (void)in_sizes; (void)n_in; (void)out_size; (void)ws_size;

    const int*           species   = (const int*)d_in[0];
    const int*           nb_spec   = (const int*)d_in[1];
    const float*         evec      = (const float*)d_in[2];
    const float*         edist     = (const float*)d_in[3];
    const unsigned char* mask      = (const unsigned char*)d_in[4];
    const int*           rev_idx   = (const int*)d_in[5];
    const float*         cutoff    = (const float*)d_in[6];
    const float* node_embed = (const float*)d_in[7];   // [L,V,D]
    const float* neigh_emb  = (const float*)d_in[8];   // [V,D]
    const float* W_edge = (const float*)d_in[9];       // [L,4,D]
    const float* b_edge = (const float*)d_in[10];
    const float* ln1_g  = (const float*)d_in[11];
    const float* ln1_b  = (const float*)d_in[12];
    const float* Wqkv   = (const float*)d_in[13];      // [L,256,768]
    const float* bqkv   = (const float*)d_in[14];
    const float* Wo     = (const float*)d_in[15];
    const float* bo     = (const float*)d_in[16];
    const float* ln2_g  = (const float*)d_in[17];
    const float* ln2_b  = (const float*)d_in[18];
    const float* Wff1   = (const float*)d_in[19];      // [L,256,1024]
    const float* bff1   = (const float*)d_in[20];
    const float* Wff2   = (const float*)d_in[21];      // [L,1024,256]
    const float* bff2   = (const float*)d_in[22];
    const float* Wnode  = (const float*)d_in[23];
    const float* bnode  = (const float*)d_in[24];
    const float* Wnh    = (const float*)d_in[25];
    const float* bnh    = (const float*)d_in[26];
    const float* Wnf    = (const float*)d_in[27];      // [L,256,1]
    const float* bnf    = (const float*)d_in[28];
    const float* Weh    = (const float*)d_in[29];
    const float* beh    = (const float*)d_in[30];
    const float* Wef    = (const float*)d_in[31];
    const float* bef    = (const float*)d_in[32];

    float* energies = (float*)d_out;

    // ---- workspace arena (manual lifetime reuse) ----
    char* ws = (char*)d_ws;
    size_t off = 0;
    auto alloc = [&](size_t bytes) -> char* {
        char* p = ws + off;
        off = (off + bytes + 255) & ~(size_t)255;
        return p;
    };
    float*  msgs   = (float*)alloc((size_t)NTOK * DM * 4);
    float*  x      = (float*)alloc((size_t)NTOK * DM * 4);
    bf16_t* B1     = (bf16_t*)alloc((size_t)NTOK * DM * 2);   // h / h2 / output_edge
    bf16_t* B2     = (bf16_t*)alloc((size_t)NTOK * DM * 2);   // attn / edge-head
    float*  big    = (float*)alloc((size_t)NTOK * 768 * 4);   // qkv f32; reused as ff1 bf16
    bf16_t* big_h  = (bf16_t*)big;
    bf16_t* WqkvT  = (bf16_t*)alloc((size_t)768 * 256 * 2);
    bf16_t* WoT    = (bf16_t*)alloc((size_t)256 * 256 * 2);
    bf16_t* Wff1T  = (bf16_t*)alloc((size_t)1024 * 256 * 2);
    bf16_t* Wff2T  = (bf16_t*)alloc((size_t)256 * 1024 * 2);
    bf16_t* WnodeT = (bf16_t*)alloc((size_t)256 * 256 * 2);
    bf16_t* WnhT   = (bf16_t*)alloc((size_t)256 * 256 * 2);
    bf16_t* WehT   = (bf16_t*)alloc((size_t)256 * 256 * 2);
    float*  nodes  = (float*)alloc((size_t)NATOM * DM * 4);
    float*  onodef = (float*)alloc((size_t)NATOM * DM * 4);
    bf16_t* onodeh = (bf16_t*)alloc((size_t)NATOM * DM * 2);
    bf16_t* nodeh  = (bf16_t*)alloc((size_t)NATOM * DM * 2);
    bf16_t* pooled = (bf16_t*)alloc((size_t)NATOM * DM * 2);

    const dim3 blk(256);
    const int gTokD = (NTOK * DM) / 256;    // 131072 blocks

    zero_out_kernel<<<dim3(8), blk, 0, stream>>>(energies, NATOM);
    msgs_init_kernel<<<dim3(gTokD), blk, 0, stream>>>(nb_spec, neigh_emb, msgs);

    for (int l = 0; l < NLAY; ++l) {
        // layer weight slices
        const float* We_l   = W_edge + (size_t)l * 4 * DM;
        const float* be_l   = b_edge + (size_t)l * DM;
        const float* g1_l   = ln1_g + (size_t)l * DM;
        const float* b1_l   = ln1_b + (size_t)l * DM;
        const float* g2_l   = ln2_g + (size_t)l * DM;
        const float* b2_l   = ln2_b + (size_t)l * DM;
        const float* Wqkv_l = Wqkv + (size_t)l * DM * 768;
        const float* Wo_l   = Wo   + (size_t)l * DM * DM;
        const float* Wff1_l = Wff1 + (size_t)l * DM * 1024;
        const float* Wff2_l = Wff2 + (size_t)l * 1024 * DM;
        const float* Wnode_l= Wnode+ (size_t)l * DM * DM;
        const float* Wnh_l  = Wnh  + (size_t)l * DM * DM;
        const float* Weh_l  = Weh  + (size_t)l * DM * DM;

        // pre-transpose weights to bf16 [N,K]
        transpose_to_bf16_kernel<<<dim3((256*768)/256), blk, 0, stream>>>(Wqkv_l, WqkvT, 256, 768);
        transpose_to_bf16_kernel<<<dim3((256*256)/256), blk, 0, stream>>>(Wo_l,   WoT,   256, 256);
        transpose_to_bf16_kernel<<<dim3((256*1024)/256), blk, 0, stream>>>(Wff1_l, Wff1T, 256, 1024);
        transpose_to_bf16_kernel<<<dim3((1024*256)/256), blk, 0, stream>>>(Wff2_l, Wff2T, 1024, 256);
        transpose_to_bf16_kernel<<<dim3((256*256)/256), blk, 0, stream>>>(Wnode_l, WnodeT, 256, 256);
        transpose_to_bf16_kernel<<<dim3((256*256)/256), blk, 0, stream>>>(Wnh_l,  WnhT,  256, 256);
        transpose_to_bf16_kernel<<<dim3((256*256)/256), blk, 0, stream>>>(Weh_l,  WehT,  256, 256);

        // x = msgs + gelu(edge) + nodes ; h = LN1(x)
        build_x_ln1_kernel<<<dim3(NTOK), blk, 0, stream>>>(
            msgs, evec, edist, species, node_embed + (size_t)l * VSP * DM,
            We_l, be_l, g1_l, b1_l, x, B1);

        // qkv = h @ Wqkv + bqkv  (f32 out)
        gemm_bf16_wmma<0, false, false><<<dim3(NTOK/128, 768/64), blk, 0, stream>>>(
            B1, WqkvT, bqkv + (size_t)l * 768, nullptr, big, nullptr, NTOK, 768, 256);

        // attention (bf16 out in B2)
        attention_kernel<<<dim3(NATOM * 2), blk, 0, stream>>>(big, mask, cutoff, B2);

        // x += attn @ Wo + bo
        gemm_bf16_wmma<0, true, false><<<dim3(NTOK/128, 256/64), blk, 0, stream>>>(
            B2, WoT, bo + (size_t)l * DM, x, x, nullptr, NTOK, 256, 256);

        // h2 = LN2(x)
        ln_kernel<<<dim3(NTOK), blk, 0, stream>>>(x, g2_l, b2_l, B1);

        // ff1 = gelu(h2 @ Wff1 + bff1)  (bf16 out, reuses qkv buffer)
        gemm_bf16_wmma<1, false, true><<<dim3(NTOK/128, 1024/64), blk, 0, stream>>>(
            B1, Wff1T, bff1 + (size_t)l * 1024, nullptr, nullptr, big_h, NTOK, 1024, 256);

        // x += ff1 @ Wff2 + bff2
        gemm_bf16_wmma<0, true, false><<<dim3(NTOK/128, 256/64), blk, 0, stream>>>(
            big_h, Wff2T, bff2 + (size_t)l * DM, x, x, nullptr, NTOK, 256, 1024);

        // output_edge (mask x in place, bf16 copy in B1), pooled
        mask_pool_kernel<<<dim3(NATOM), blk, 0, stream>>>(x, mask, cutoff, B1, pooled);

        // output_node = nodes + pooled @ Wnode + bnode
        nodes_gather_kernel<<<dim3((NATOM*DM)/256), blk, 0, stream>>>(
            species, node_embed + (size_t)l * VSP * DM, nodes);
        gemm_bf16_wmma<0, true, false><<<dim3(NATOM/128, 256/64), blk, 0, stream>>>(
            pooled, WnodeT, bnode + (size_t)l * DM, nodes, onodef, nullptr, NATOM, 256, 256);
        f32_to_bf16_kernel<<<dim3((NATOM*DM)/256), blk, 0, stream>>>(onodef, onodeh, NATOM * DM);

        // node energy head
        gemm_bf16_wmma<1, false, true><<<dim3(NATOM/128, 256/64), blk, 0, stream>>>(
            onodeh, WnhT, bnh + (size_t)l * DM, nullptr, nullptr, nodeh, NATOM, 256, 256);
        node_energy_kernel<<<dim3(NATOM/8), blk, 0, stream>>>(
            nodeh, Wnf + (size_t)l * DM, bnf + l, energies);

        // edge energy head
        gemm_bf16_wmma<1, false, true><<<dim3(NTOK/128, 256/64), blk, 0, stream>>>(
            B1, WehT, beh + (size_t)l * DM, nullptr, nullptr, B2, NTOK, 256, 256);
        edge_energy_kernel<<<dim3(NTOK/8), blk, 0, stream>>>(
            B2, Wef + (size_t)l * DM, bef + l, mask, cutoff, energies);

        // reverse-message mixing
        rev_mix_kernel<<<dim3(gTokD), blk, 0, stream>>>(msgs, x, rev_idx, mask);
    }
}
